// Qwen3MoeSparseMoeBlock_2551210574776
// MI455X (gfx1250) — matrix operand order repaired and checked by reading, not compiled, not run on hardware
//
#include <hip/hip_runtime.h>

// ---------------------------------------------------------------------------
// Qwen3 MoE sparse block for gfx1250 (MI455X), bf16 WMMA formulation.
// T=1024 tokens, H=2048, I=768, E=64 experts, top-K=8.
// Double-buffered pipeline: async A->LDS + register-staged B overlap WMMA;
// B fragments rotate one-ahead so ds loads stay in flight under WMMA.
// ---------------------------------------------------------------------------

#define H_ 2048
#define I_ 768
#define E_ 64
#define T_ 1024
#define K_ 8

typedef __attribute__((ext_vector_type(16))) __bf16 v16bf;
typedef __attribute__((ext_vector_type(8)))  __bf16 v8bf;
typedef __attribute__((ext_vector_type(8)))  float  v8f;

static __device__ __forceinline__ v16bf join16(const __bf16* p0, const __bf16* p1) {
  v8bf lo = *(const v8bf*)p0;
  v8bf hi = *(const v8bf*)p1;
  return __builtin_shufflevector(lo, hi, 0,1,2,3,4,5,6,7,8,9,10,11,12,13,14,15);
}

// LDS byte offset of a generic pointer to __shared__ (aperture rule: addr[31:0]).
static __device__ __forceinline__ unsigned lds_off(const void* p) {
  return (unsigned)(uintptr_t)p;
}

// Async 16-byte gather global->LDS (GLOBAL_LOAD_ASYNC_TO_LDS_B128, ASYNCcnt).
static __device__ __forceinline__ void async_ld128(unsigned lds, const void* g) {
  asm volatile("global_load_async_to_lds_b128 %0, %1, off"
               :: "v"(lds), "v"(g) : "memory");
}
static __device__ __forceinline__ void wait_async0() {
  asm volatile("s_wait_asynccnt 0" ::: "memory");
}

// ---------------------------------------------------------------------------
// 0) zero output + counts
// ---------------------------------------------------------------------------
__global__ void moe_init(float* __restrict__ out, int n, int* __restrict__ counts) {
  int i = blockIdx.x * blockDim.x + threadIdx.x;
  if (i < n) out[i] = 0.0f;
  if (i < E_) counts[i] = 0;
}

// ---------------------------------------------------------------------------
// 0b) one-shot fp32 -> bf16 copy of hidden states (feeds async A staging)
// ---------------------------------------------------------------------------
__global__ void moe_cvt_x(const float* __restrict__ x, __bf16* __restrict__ xb, int n) {
  int i = (blockIdx.x * blockDim.x + threadIdx.x) * 4;
  if (i < n) {
    float4 f = *(const float4*)(x + i);
    xb[i + 0] = (__bf16)f.x;
    xb[i + 1] = (__bf16)f.y;
    xb[i + 2] = (__bf16)f.z;
    xb[i + 3] = (__bf16)f.w;
  }
}

// ---------------------------------------------------------------------------
// 1) router: logits -> softmax(fp32) -> top-8 -> renormalize
// ---------------------------------------------------------------------------
__global__ __launch_bounds__(64) void moe_router(
    const float* __restrict__ x, const float* __restrict__ gw,
    int* __restrict__ counts, int* __restrict__ topk_e, float* __restrict__ topk_w) {
  const int t = blockIdx.x;
  const int e = threadIdx.x;
  const float* xr = x + (size_t)t * H_;
  const float* wr = gw + (size_t)e * H_;
  float d = 0.0f;
  #pragma unroll 4
  for (int k = 0; k < H_; k += 4) {
    float4 a = *(const float4*)(xr + k);
    float4 b = *(const float4*)(wr + k);
    d += a.x * b.x + a.y * b.y + a.z * b.z + a.w * b.w;
  }
  __shared__ float lg[E_];
  lg[e] = d;
  __syncthreads();
  if (e == 0) {
    float mx = lg[0];
    for (int i = 1; i < E_; ++i) mx = fmaxf(mx, lg[i]);
    float p[E_]; float s = 0.0f;
    for (int i = 0; i < E_; ++i) { p[i] = __expf(lg[i] - mx); s += p[i]; }
    float inv_s = 1.0f / s;
    int   sel[K_]; float selw[K_]; float wsum = 0.0f;
    for (int k = 0; k < K_; ++k) {
      int best = 0; float bv = -1.0f;
      for (int i = 0; i < E_; ++i) { if (p[i] > bv) { bv = p[i]; best = i; } }
      sel[k] = best; selw[k] = bv * inv_s; wsum += selw[k];
      p[best] = -1.0f;
    }
    float invw = (wsum == 0.0f) ? 1.0f : (1.0f / wsum);
    for (int k = 0; k < K_; ++k) {
      topk_e[t * K_ + k] = sel[k];
      topk_w[t * K_ + k] = selw[k] * invw;
      atomicAdd(&counts[sel[k]], 1);
    }
  }
}

// ---------------------------------------------------------------------------
// 2) exclusive scan over 64 counts; reset scatter cursors
// ---------------------------------------------------------------------------
__global__ __launch_bounds__(64) void moe_scan(
    const int* __restrict__ counts, int* __restrict__ offsets, int* __restrict__ cursors) {
  if (threadIdx.x == 0) {
    int s = 0;
    for (int e = 0; e < E_; ++e) { offsets[e] = s; s += counts[e]; }
    offsets[E_] = s;
  }
  if (threadIdx.x < E_) cursors[threadIdx.x] = 0;
}

// ---------------------------------------------------------------------------
// 3) scatter (token, weight) pairs into per-expert compact row lists
// ---------------------------------------------------------------------------
__global__ void moe_scatter(
    const int* __restrict__ topk_e, const float* __restrict__ topk_w,
    const int* __restrict__ offsets, int* __restrict__ cursors,
    int* __restrict__ rows_token, float* __restrict__ rows_w) {
  int i = blockIdx.x * blockDim.x + threadIdx.x;
  if (i >= T_ * K_) return;
  int e = topk_e[i];
  int pos = atomicAdd(&cursors[e], 1);
  int slot = offsets[e] + pos;
  rows_token[slot] = i >> 3;
  rows_w[slot] = topk_w[i];
}

// ---------------------------------------------------------------------------
// 4) gate_up GEMM + SwiGLU.  Per block: expert e, 128 gathered tokens (M),
//    64 gate + 64 up cols (N).  Double-buffered pipeline.
// ---------------------------------------------------------------------------
__global__ __launch_bounds__(256, 1) void moe_gate_up(
    const __bf16* __restrict__ xb, const float* __restrict__ wgu,
    const int* __restrict__ counts, const int* __restrict__ offsets,
    const int* __restrict__ rows_token, __bf16* __restrict__ act) {
  const int e  = blockIdx.z;
  const int mt = blockIdx.y;
  const int cnt = counts[e];
  if (mt * 128 >= cnt) return;          // block-uniform exit: EXEC stays full
  const int nt = blockIdx.x;            // 0..11  -> col0 = nt*64
  const int rowBase = offsets[e];
  const int col0 = nt * 64;
  const float* W = wgu + (size_t)e * H_ * (2 * I_);

  __shared__ __bf16 Ash[2][128][72];
  __shared__ __bf16 Bg[2][64][72];
  __shared__ __bf16 Bu[2][64][72];

  const int tid  = threadIdx.x;
  const int lane = tid & 31;
  const int wv   = tid >> 5;      // wave 0..7
  const int grp  = lane >> 4;     // 0/1
  const int l15  = lane & 15;

  v8f accG[4] = {}; v8f accU[4] = {};

  // A-staging: thread t covers row t/2, 32 bf16 starting at (t&1)*32
  const int ar = tid >> 1;
  const int ah = (tid & 1) * 32;
  const int arow = mt * 128 + ar;
  const bool avalid = arow < cnt;
  const __bf16* xrow =
      avalid ? (xb + (size_t)rows_token[rowBase + arow] * H_) : xb;
  // B-staging: thread t covers col t&63, K phases t>>6 stepping 4
  const int bn = tid & 63;
  const int bk = tid >> 6;

  auto stageA = [&](int k0, int buf) {
    if (avalid) {
      const __bf16* g = xrow + k0 + ah;
      const unsigned l = lds_off(&Ash[buf][ar][ah]);
      async_ld128(l +  0, g +  0);
      async_ld128(l + 16, g +  8);
      async_ld128(l + 32, g + 16);
      async_ld128(l + 48, g + 24);
    } else {
      v8bf z = {};
      #pragma unroll
      for (int j = 0; j < 32; j += 8) *(v8bf*)&Ash[buf][ar][ah + j] = z;
    }
  };
  auto issueB = [&](int k0, float* bg, float* bu) {
    #pragma unroll
    for (int k = 0; k < 64; k += 4) {
      const size_t rb = (size_t)(k0 + k + bk) * (2 * I_);
      bg[k >> 2] = W[rb + col0 + bn];
      bu[k >> 2] = W[rb + I_ + col0 + bn];
    }
  };
  auto commitB = [&](int buf, const float* bg, const float* bu) {
    #pragma unroll
    for (int k = 0; k < 64; k += 4) {
      const int kk = k + bk;
      Bg[buf][bn][kk] = (__bf16)bg[k >> 2];
      Bu[buf][bn][kk] = (__bf16)bu[k >> 2];
    }
  };
  auto compute = [&](int buf) {
    #pragma unroll
    for (int ks = 0; ks < 64; ks += 32) {
      const int row = 16 * wv + l15;
      v16bf a = join16(&Ash[buf][row][ks + grp * 8],
                       &Ash[buf][row][ks + 16 + grp * 8]);
      // 8-fragment stream: even t -> gate frag t/2, odd t -> up frag t/2.
      // One-ahead rotation keeps a ds_load pair in flight under each WMMA.
      auto loadFrag = [&](int t) -> v16bf {
        const int n = (t >> 1) * 16 + l15;
        const __bf16* base = ((t & 1) == 0) ? &Bg[buf][n][ks + grp * 8]
                                            : &Bu[buf][n][ks + grp * 8];
        return join16(base, base + 16);
      };
      v16bf bcur = loadFrag(0);
      #pragma unroll
      for (int t = 0; t < 8; ++t) {
        v16bf bnext;
        if (t < 7) bnext = loadFrag(t + 1);
        if ((t & 1) == 0)
          accG[t >> 1] = __builtin_amdgcn_wmma_f32_16x16x32_bf16(
              false, a, false, bcur, (short)0, accG[t >> 1], false, false);
        else
          accU[t >> 1] = __builtin_amdgcn_wmma_f32_16x16x32_bf16(
              false, a, false, bcur, (short)0, accU[t >> 1], false, false);
        if (t < 7) bcur = bnext;
      }
    }
  };

  // prologue: fill buffer 0
  float bgr[16], bur[16];
  stageA(0, 0);
  issueB(0, bgr, bur);
  commitB(0, bgr, bur);
  wait_async0();
  __syncthreads();

  int cur = 0;
  for (int k0 = 0; k0 < H_; k0 += 64) {
    const int nxt = cur ^ 1;
    const bool has_next = (k0 + 64) < H_;
    if (has_next) {
      stageA(k0 + 64, nxt);          // ASYNCcnt, lands during compute
      issueB(k0 + 64, bgr, bur);     // LOADcnt, lands during compute
    }
    compute(cur);
    if (has_next) commitB(nxt, bgr, bur);
    wait_async0();
    __syncthreads();
    cur = nxt;
  }

  // SwiGLU + store act (bf16). C layout: VGPR r -> M = r + 8*grp, N = l15.
  #pragma unroll
  for (int t4 = 0; t4 < 4; ++t4) {
    #pragma unroll
    for (int r = 0; r < 8; ++r) {
      const int mlocal = mt * 128 + 16 * wv + 8 * grp + r;
      if (mlocal < cnt) {
        float g = accG[t4][r], u = accU[t4][r];
        float s = g / (1.0f + __expf(-g));
        act[(size_t)(rowBase + mlocal) * I_ + col0 + t4 * 16 + l15] = (__bf16)(s * u);
      }
    }
  }
}

// ---------------------------------------------------------------------------
// 5) down GEMM: act(bf16, async-staged) x Wd(fp32->bf16), weighted atomics.
//    Per block: expert e, 128 gathered rows (M), 128 H-cols (N).
// ---------------------------------------------------------------------------
__global__ __launch_bounds__(256, 1) void moe_down(
    const __bf16* __restrict__ act, const float* __restrict__ wd,
    const int* __restrict__ counts, const int* __restrict__ offsets,
    const int* __restrict__ rows_token, const float* __restrict__ rows_w,
    float* __restrict__ out) {
  const int e  = blockIdx.z;
  const int mt = blockIdx.y;
  const int cnt = counts[e];
  if (mt * 128 >= cnt) return;
  const int nt = blockIdx.x;            // 0..15 -> n0 = nt*128
  const int rowBase = offsets[e];
  const int n0 = nt * 128;
  const float* W = wd + (size_t)e * I_ * H_;

  __shared__ __bf16 Ash[2][128][72];
  __shared__ __bf16 Bsh[2][128][72];

  const int tid  = threadIdx.x;
  const int lane = tid & 31;
  const int wv   = tid >> 5;
  const int grp  = lane >> 4;
  const int l15  = lane & 15;

  v8f acc[8] = {};

  const int ar = tid >> 1;
  const int ah = (tid & 1) * 32;
  const int arow = mt * 128 + ar;
  const bool avalid = arow < cnt;
  const __bf16* actrow = act + (size_t)(rowBase + (avalid ? arow : 0)) * I_;
  const int bn = tid & 127;
  const int bk = tid >> 7;

  auto stageA = [&](int k0, int buf) {
    if (avalid) {
      const __bf16* g = actrow + k0 + ah;
      const unsigned l = lds_off(&Ash[buf][ar][ah]);
      async_ld128(l +  0, g +  0);
      async_ld128(l + 16, g +  8);
      async_ld128(l + 32, g + 16);
      async_ld128(l + 48, g + 24);
    } else {
      v8bf z = {};
      #pragma unroll
      for (int j = 0; j < 32; j += 8) *(v8bf*)&Ash[buf][ar][ah + j] = z;
    }
  };
  auto issueB = [&](int k0, float* br) {
    #pragma unroll
    for (int k = 0; k < 64; k += 2)
      br[k >> 1] = W[(size_t)(k0 + k + bk) * H_ + n0 + bn];
  };
  auto commitB = [&](int buf, const float* br) {
    #pragma unroll
    for (int k = 0; k < 64; k += 2)
      Bsh[buf][bn][k + bk] = (__bf16)br[k >> 1];
  };
  auto compute = [&](int buf) {
    #pragma unroll
    for (int ks = 0; ks < 64; ks += 32) {
      const int row = 16 * wv + l15;
      v16bf a = join16(&Ash[buf][row][ks + grp * 8],
                       &Ash[buf][row][ks + 16 + grp * 8]);
      auto loadFrag = [&](int t) -> v16bf {
        const __bf16* base = &Bsh[buf][t * 16 + l15][ks + grp * 8];
        return join16(base, base + 16);
      };
      v16bf bcur = loadFrag(0);
      #pragma unroll
      for (int t = 0; t < 8; ++t) {
        v16bf bnext;
        if (t < 7) bnext = loadFrag(t + 1);
        acc[t] = __builtin_amdgcn_wmma_f32_16x16x32_bf16(
            false, a, false, bcur, (short)0, acc[t], false, false);
        if (t < 7) bcur = bnext;
      }
    }
  };

  float br[32];
  stageA(0, 0);
  issueB(0, br);
  commitB(0, br);
  wait_async0();
  __syncthreads();

  int cur = 0;
  for (int k0 = 0; k0 < I_; k0 += 64) {
    const int nxt = cur ^ 1;
    const bool has_next = (k0 + 64) < I_;
    if (has_next) {
      stageA(k0 + 64, nxt);
      issueB(k0 + 64, br);
    }
    compute(cur);
    if (has_next) commitB(nxt, br);
    wait_async0();
    __syncthreads();
    cur = nxt;
  }

  // epilogue: weighted atomic accumulation into out[token][h]
  int   tok[8]; float rwv[8]; bool ok[8];
  #pragma unroll
  for (int r = 0; r < 8; ++r) {
    const int mlocal = mt * 128 + 16 * wv + 8 * grp + r;
    ok[r] = mlocal < cnt;
    const int slot = rowBase + (ok[r] ? mlocal : 0);
    tok[r] = rows_token[slot];
    rwv[r] = rows_w[slot];
  }
  #pragma unroll
  for (int t8 = 0; t8 < 8; ++t8) {
    #pragma unroll
    for (int r = 0; r < 8; ++r) {
      if (ok[r])
        atomicAdd(&out[(size_t)tok[r] * H_ + n0 + t8 * 16 + l15], rwv[r] * acc[t8][r]);
    }
  }
}

// ---------------------------------------------------------------------------
// launch
// ---------------------------------------------------------------------------
extern "C" void kernel_launch(void* const* d_in, const int* in_sizes, int n_in,
                              void* d_out, int out_size, void* d_ws, size_t ws_size,
                              hipStream_t stream) {
  (void)in_sizes; (void)n_in; (void)out_size; (void)ws_size;
  const float* x   = (const float*)d_in[0];   // [T, H]
  const float* gw  = (const float*)d_in[1];   // [E, H]
  const float* wgu = (const float*)d_in[2];   // [E, H, 2I]
  const float* wd  = (const float*)d_in[3];   // [E, I, H]
  float* out = (float*)d_out;                 // [T, H]

  char* w = (char*)d_ws;
  int*    counts     = (int*)w;    w += 256;
  int*    cursors    = (int*)w;    w += 256;
  int*    offsets    = (int*)w;    w += 512;               // 65 ints, padded
  int*    topk_e     = (int*)w;    w += (size_t)T_ * K_ * 4;
  float*  topk_w     = (float*)w;  w += (size_t)T_ * K_ * 4;
  int*    rows_token = (int*)w;    w += (size_t)T_ * K_ * 4;
  float*  rows_w     = (float*)w;  w += (size_t)T_ * K_ * 4;
  __bf16* act        = (__bf16*)w; w += (size_t)T_ * K_ * I_ * 2; // 12.6MB
  __bf16* xb         = (__bf16*)w;                                // [T,H] 4MB

  const int nOut = T_ * H_;
  moe_init<<<(nOut + 255) / 256, 256, 0, stream>>>(out, nOut, counts);
  moe_cvt_x<<<(T_ * H_ / 4 + 255) / 256, 256, 0, stream>>>(x, xb, T_ * H_);
  moe_router<<<T_, 64, 0, stream>>>(x, gw, counts, topk_e, topk_w);
  moe_scan<<<1, 64, 0, stream>>>(counts, offsets, cursors);
  moe_scatter<<<(T_ * K_ + 255) / 256, 256, 0, stream>>>(
      topk_e, topk_w, offsets, cursors, rows_token, rows_w);
  moe_gate_up<<<dim3(I_ / 64, 8, E_), 256, 0, stream>>>(
      xb, wgu, counts, offsets, rows_token, act);
  moe_down<<<dim3(H_ / 128, 8, E_), 256, 0, stream>>>(
      act, wd, counts, offsets, rows_token, rows_w, out);
}
